// KDABlock_91302414778575
// MI455X (gfx1250) — compile-verified, matrix operand order checked by reading, not run on hardware
//
#include <hip/hip_runtime.h>
#include <cstdint>

// ---------------- problem constants ----------------
#define BB     2
#define TT     4096
#define DD     1024
#define HH     8
#define DKK    64
#define DVV    64
#define RGG    64
#define DFFF   2048
#define KCONV  4
#define MM     (BB * TT)          // 8192 token rows
#define DECAYC 0.9f
#define THC    0.5f
#define AFLOORC 1e-4f

typedef __attribute__((ext_vector_type(16))) _Float16 v16h;
typedef __attribute__((ext_vector_type(8)))  _Float16 v8h;
typedef __attribute__((ext_vector_type(8)))  float    v8f;

__device__ __forceinline__ float sigf(float x) { return 1.f / (1.f + __expf(-x)); }
__device__ __forceinline__ float siluf(float x) { return x * sigf(x); }

// ---------------- fp32 -> fp16 convert (grid-stride) ----------------
__global__ void f32_to_f16_kernel(const float* __restrict__ in,
                                  _Float16* __restrict__ out, size_t n) {
  for (size_t i = (size_t)blockIdx.x * blockDim.x + threadIdx.x; i < n;
       i += (size_t)gridDim.x * blockDim.x)
    out[i] = (_Float16)in[i];
}

// ---------------- row RMSNorm (width Dw), fp32 and/or fp16 out -------------
__global__ void rms_row_kernel(const float* __restrict__ in,
                               const float* __restrict__ w,
                               float* __restrict__ out32,
                               _Float16* __restrict__ out16, int Dw) {
  size_t row = blockIdx.x;
  const float* x = in + row * (size_t)Dw;
  __shared__ float red[256];
  float ss = 0.f;
  for (int d = threadIdx.x; d < Dw; d += 256) { float v = x[d]; ss += v * v; }
  red[threadIdx.x] = ss;
  __syncthreads();
  for (int s = 128; s > 0; s >>= 1) {
    if ((int)threadIdx.x < s) red[threadIdx.x] += red[threadIdx.x + s];
    __syncthreads();
  }
  float scale = rsqrtf(red[0] / (float)Dw + 1e-6f);
  for (int d = threadIdx.x; d < Dw; d += 256) {
    float v = x[d] * scale * w[d];
    if (out32) out32[row * (size_t)Dw + d] = v;
    if (out16) out16[row * (size_t)Dw + d] = (_Float16)v;
  }
}

// ---------------- causal depthwise conv (K=4) + SiLU -> fp16 ---------------
__global__ void conv_silu_kernel(const float* __restrict__ hn,
                                 const float* __restrict__ cw,
                                 const float* __restrict__ cb,
                                 _Float16* __restrict__ h16) {
  size_t row = blockIdx.x;
  int b = (int)(row / TT), t = (int)(row % TT);
  for (int d = threadIdx.x; d < DD; d += 256) {
    float acc = cb[d];
#pragma unroll
    for (int j = 0; j < KCONV; ++j) {
      int tt = t + j - (KCONV - 1);
      if (tt >= 0) acc += hn[((size_t)b * TT + tt) * DD + d] * cw[d * KCONV + j];
    }
    h16[row * DD + d] = (_Float16)siluf(acc);
  }
}

// ---------------- WMMA fp16 GEMM:  C[M,N] = A[M,K] * W[N,K]^T + bias --------
// act: 0=none 1=silu 2=sigmoid.  Optional fp16 mirror Ch.
// Each wave owns a 32x32 macro-tile (2x2 WMMA accumulators): 4 WMMAs per
// 8 x b128 fragment loads -> 2x the FLOP/byte of a single-tile wave.
// Block = 128 threads (4 waves) -> 64x64 block tile. grid(N/64, M/64).
__device__ __forceinline__ v16h load_frag(const _Float16* __restrict__ rowp,
                                          int k0, int lane) {
  int half = lane >> 4;
  const _Float16* p = rowp + k0 + half * 8;
  v8h lo = *(const v8h*)(p);
  v8h hi = *(const v8h*)(p + 16);
  v16h f;
#pragma unroll
  for (int i = 0; i < 8; ++i) { f[i] = lo[i]; f[i + 8] = hi[i]; }
  return f;
}

__device__ __forceinline__ void store_tile(v8f acc, int m0, int n0, int lane,
                                           const float* __restrict__ bias,
                                           float* __restrict__ Cf,
                                           _Float16* __restrict__ Ch,
                                           int Nn, int act) {
  int r = lane & 15;
  int half = lane >> 4;
#pragma unroll
  for (int i = 0; i < 8; ++i) {
    int row = m0 + i + half * 8;
    int col = n0 + r;
    float val = acc[i] + (bias ? bias[col] : 0.f);
    if (act == 1) val = siluf(val);
    else if (act == 2) val = sigf(val);
    Cf[(size_t)row * Nn + col] = val;
    if (Ch) Ch[(size_t)row * Nn + col] = (_Float16)val;
  }
}

__global__ __launch_bounds__(128) void gemm_wmma_kernel(
    const _Float16* __restrict__ A, const _Float16* __restrict__ W,
    const float* __restrict__ bias, float* __restrict__ Cf,
    _Float16* __restrict__ Ch, int Mn, int Nn, int Kn, int act) {
  int lane = threadIdx.x & 31;
  int wv = threadIdx.x >> 5;
  int m0 = blockIdx.y * 64 + (wv >> 1) * 32;
  int n0 = blockIdx.x * 64 + (wv & 1) * 32;
  int r = lane & 15;
  const _Float16* pa0 = A + (size_t)(m0 + r) * Kn;
  const _Float16* pa1 = A + (size_t)(m0 + 16 + r) * Kn;
  const _Float16* pb0 = W + (size_t)(n0 + r) * Kn;
  const _Float16* pb1 = W + (size_t)(n0 + 16 + r) * Kn;
  v8f a00 = {}, a01 = {}, a10 = {}, a11 = {};
  for (int k0 = 0; k0 < Kn; k0 += 32) {
    v16h fa0 = load_frag(pa0, k0, lane);
    v16h fa1 = load_frag(pa1, k0, lane);
    v16h fb0 = load_frag(pb0, k0, lane);
    v16h fb1 = load_frag(pb1, k0, lane);
    a00 = __builtin_amdgcn_wmma_f32_16x16x32_f16(false, fa0, false, fb0,
                                                 (short)0, a00, false, false);
    a01 = __builtin_amdgcn_wmma_f32_16x16x32_f16(false, fa0, false, fb1,
                                                 (short)0, a01, false, false);
    a10 = __builtin_amdgcn_wmma_f32_16x16x32_f16(false, fa1, false, fb0,
                                                 (short)0, a10, false, false);
    a11 = __builtin_amdgcn_wmma_f32_16x16x32_f16(false, fa1, false, fb1,
                                                 (short)0, a11, false, false);
  }
  store_tile(a00, m0,      n0,      lane, bias, Cf, Ch, Nn, act);
  store_tile(a01, m0,      n0 + 16, lane, bias, Cf, Ch, Nn, act);
  store_tile(a10, m0 + 16, n0,      lane, bias, Cf, Ch, Nn, act);
  store_tile(a11, m0 + 16, n0 + 16, lane, bias, Cf, Ch, Nn, act);
}

// ---------------- per-head RMS on q,k,v (in-place, 64 wide) ----------------
__global__ void qkv_rms_kernel(float* __restrict__ qkvs,
                               const float* __restrict__ qw,
                               const float* __restrict__ kw,
                               const float* __restrict__ vw) {
  int idx = blockIdx.x;
  int which = idx / (MM * HH);
  int rem = idx % (MM * HH);
  size_t row = rem / HH;
  int h = rem % HH;
  float* p = qkvs + row * 2048 + which * 512 + h * 64;
  const float* w = (which == 0 ? qw : (which == 1 ? kw : vw)) + h * 64;
  __shared__ float red[64];
  int tid = threadIdx.x;
  float v = p[tid];
  red[tid] = v * v;
  __syncthreads();
  if (tid == 0) {
    float s = 0.f;
    for (int i = 0; i < 64; ++i) s += red[i];
    red[0] = rsqrtf(s / 64.f + 1e-6f);
  }
  __syncthreads();
  p[tid] = v * red[0] * w[tid];
}

// ---------------- per-head RMS on y -> fp16 --------------------------------
__global__ void head_rms_kernel(const float* __restrict__ y,
                                const float* __restrict__ w,
                                _Float16* __restrict__ out) {
  int idx = blockIdx.x;
  size_t row = idx / HH;
  int h = idx % HH;
  const float* p = y + row * 512 + h * 64;
  __shared__ float red[64];
  int tid = threadIdx.x;
  float v = p[tid];
  red[tid] = v * v;
  __syncthreads();
  if (tid == 0) {
    float s = 0.f;
    for (int i = 0; i < 64; ++i) s += red[i];
    red[0] = rsqrtf(s / 64.f + 1e-6f);
  }
  __syncthreads();
  out[row * 512 + h * 64 + tid] = (_Float16)(v * red[0] * w[h * 64 + tid]);
}

// ---------------- beta projection (N=8, plain dot) -------------------------
__global__ void beta_kernel(const _Float16* __restrict__ h16,
                            const float* __restrict__ bw,
                            const float* __restrict__ bb,
                            float* __restrict__ out) {
  int idx = blockIdx.x * blockDim.x + threadIdx.x;
  if (idx >= MM * HH) return;
  size_t row = idx / HH;
  int h = idx % HH;
  float acc = bb[h];
  const _Float16* hp = h16 + row * DD;
  const float* wp = bw + (size_t)h * DD;
  for (int d = 0; d < DD; ++d) acc += (float)hp[d] * wp[d];
  out[row * 8 + h] = acc;
}

// ---------------- LIF spike chain + gate precompute ------------------------
// One block per (b,h): 64 channel threads walk T sequentially.
// Forward value of straight-through spike == hard threshold.
__global__ void spike_kernel(const float* __restrict__ qkvs,
                             float* __restrict__ alpha_io,   // in: base, out: gate
                             const float* __restrict__ bbase,
                             const float* __restrict__ a_sp,
                             const float* __restrict__ b_sp,
                             float* __restrict__ betau) {
  int bh = blockIdx.x;
  int b = bh >> 3, h = bh & 7;
  int k = threadIdx.x;  // 0..63
  __shared__ float rs[64], rc[64];
  float asp = a_sp[h * 64 + k], bsp = b_sp[h * 64 + k];
  float mem = 0.f;
  for (int t = 0; t < TT; ++t) {
    size_t row = (size_t)b * TT + t;
    float dt = qkvs[row * 2048 + 1536 + h * 64 + k];
    mem = DECAYC * mem + dt;
    float spike = (mem > THC) ? 1.f : 0.f;
    mem *= (1.f - spike);
    float al = sigf(alpha_io[row * 512 + h * 64 + k] + asp * spike);
    al = fminf(fmaxf(al, AFLOORC), 1.f);
    alpha_io[row * 512 + h * 64 + k] = al;
    rs[k] = spike * bsp;
    rc[k] = spike;
    __syncthreads();
    if (k == 0) {
      float sb = 0.f, cnt = 0.f;
      for (int i = 0; i < 64; ++i) { sb += rs[i]; cnt += rc[i]; }
      float be = fminf(fmaxf(sigf(bbase[row * 8 + h] + sb), 0.f), 1.f);
      betau[row * 8 + h] = (cnt > 0.f ? 1.f : 0.f) * be;
    }
    __syncthreads();
  }
}

// ---------------- gated delta-rule scan: state 64x64 in LDS ----------------
// One block per (b,h); 256 threads = 64 v-cols x 4 k-groups.
__global__ __launch_bounds__(256) void scan_kernel(
    const float* __restrict__ qkvs, const float* __restrict__ alphag,
    const float* __restrict__ betau, float* __restrict__ y) {
  int bh = blockIdx.x;
  int b = bh >> 3, h = bh & 7;
  __shared__ float s[DKK * DVV];
  __shared__ float kv[DKK], qv[DKK], vv[DVV], av[DKK];
  __shared__ float red[4 * DVV], pred[DVV];
  __shared__ float uS;
  int tid = threadIdx.x;
  int v = tid & 63, kg = tid >> 6;
  for (int i = tid; i < DKK * DVV; i += 256) s[i] = 0.f;
  __syncthreads();
  for (int t = 0; t < TT; ++t) {
    size_t row = (size_t)b * TT + t;
    if (tid < 64)        kv[tid]       = qkvs[row * 2048 + 512  + h * 64 + tid];
    else if (tid < 128)  qv[tid - 64]  = qkvs[row * 2048 + 0    + h * 64 + (tid - 64)];
    else if (tid < 192)  vv[tid - 128] = qkvs[row * 2048 + 1024 + h * 64 + (tid - 128)];
    else                 av[tid - 192] = alphag[row * 512 + h * 64 + (tid - 192)];
    if (tid == 0) uS = betau[row * 8 + h];
    __syncthreads();
    float p = 0.f;
#pragma unroll
    for (int j = 0; j < 16; ++j) {
      int k = kg * 16 + j;
      p += av[k] * kv[k] * s[k * 64 + v];
    }
    red[kg * 64 + v] = p;
    __syncthreads();
    if (tid < 64) pred[tid] = red[tid] + red[64 + tid] + red[128 + tid] + red[192 + tid];
    __syncthreads();
    float dv = uS * (vv[v] - pred[v]);
    float po = 0.f;
#pragma unroll
    for (int j = 0; j < 16; ++j) {
      int k = kg * 16 + j;
      float sn = av[k] * s[k * 64 + v] + kv[k] * dv;
      s[k * 64 + v] = sn;
      po += qv[k] * sn;
    }
    red[kg * 64 + v] = po;
    __syncthreads();
    if (tid < 64)
      y[row * 512 + h * 64 + tid] =
          red[tid] + red[64 + tid] + red[128 + tid] + red[192 + tid];
    __syncthreads();
  }
}

// ---------------- elementwise fuses ----------------------------------------
__global__ void fuse_gate_kernel(const float* __restrict__ x,
                                 const float* __restrict__ attn,
                                 const float* __restrict__ g,
                                 float* __restrict__ y1, size_t n) {
  for (size_t i = (size_t)blockIdx.x * 256 + threadIdx.x; i < n;
       i += (size_t)gridDim.x * 256)
    y1[i] = x[i] + attn[i] * g[i];
}

__global__ void fuse_ff_kernel(const float* __restrict__ f1,
                               const float* __restrict__ f3,
                               _Float16* __restrict__ out, size_t n) {
  for (size_t i = (size_t)blockIdx.x * 256 + threadIdx.x; i < n;
       i += (size_t)gridDim.x * 256)
    out[i] = (_Float16)(f1[i] * f3[i]);  // f1 already SiLU'd in GEMM epilogue
}

__global__ void final_add_kernel(const float* __restrict__ y1,
                                 const float* __restrict__ f2,
                                 float* __restrict__ out, size_t n) {
  for (size_t i = (size_t)blockIdx.x * 256 + threadIdx.x; i < n;
       i += (size_t)gridDim.x * 256)
    out[i] = y1[i] + f2[i];
}

// ============================ host launcher ================================
extern "C" void kernel_launch(void* const* d_in, const int* in_sizes, int n_in,
                              void* d_out, int out_size, void* d_ws, size_t ws_size,
                              hipStream_t stream) {
  (void)in_sizes; (void)n_in; (void)out_size; (void)ws_size;
  const float* x          = (const float*)d_in[0];
  const float* norm_in_w  = (const float*)d_in[1];
  const float* conv_w     = (const float*)d_in[2];
  const float* conv_b     = (const float*)d_in[3];
  const float* q_w        = (const float*)d_in[4];
  const float* k_w        = (const float*)d_in[5];
  const float* v_w        = (const float*)d_in[6];
  const float* out_w      = (const float*)d_in[7];
  const float* spike_w    = (const float*)d_in[8];
  const float* alpha_up_w = (const float*)d_in[9];
  const float* alpha_up_b = (const float*)d_in[10];
  const float* alpha_dn_w = (const float*)d_in[11];
  const float* alpha_dn_b = (const float*)d_in[12];
  const float* alpha_sp   = (const float*)d_in[13];
  const float* beta_w     = (const float*)d_in[14];
  const float* beta_b     = (const float*)d_in[15];
  const float* beta_sp    = (const float*)d_in[16];
  const float* head_nw    = (const float*)d_in[17];
  const float* q_nw       = (const float*)d_in[18];
  const float* k_nw       = (const float*)d_in[19];
  const float* v_nw       = (const float*)d_in[20];
  const float* u1_w       = (const float*)d_in[21];
  const float* u1_b       = (const float*)d_in[22];
  const float* u2_w       = (const float*)d_in[23];
  const float* u2_b       = (const float*)d_in[24];
  const float* ff_norm_w  = (const float*)d_in[25];
  const float* ff_w1      = (const float*)d_in[26];
  const float* ff_w3      = (const float*)d_in[27];
  const float* ff_w2      = (const float*)d_in[28];
  float* outp = (float*)d_out;

  // ---- workspace layout (bump allocator, 256B aligned) ----
  char* base = (char*)d_ws;
  size_t off = 0;
  auto alloc = [&](size_t bytes) -> char* {
    char* p = base + off;
    off = (off + bytes + 255) & ~(size_t)255;
    return p;
  };
  _Float16* Wqkvs = (_Float16*)alloc((size_t)2048 * 1024 * 2); // q|k|v|spike stacked
  _Float16* Wau   = (_Float16*)alloc((size_t)64 * 1024 * 2);
  _Float16* Wad   = (_Float16*)alloc((size_t)512 * 64 * 2);
  _Float16* Wout  = (_Float16*)alloc((size_t)1024 * 512 * 2);
  _Float16* Wu1   = (_Float16*)alloc((size_t)64 * 1024 * 2);
  _Float16* Wu2   = (_Float16*)alloc((size_t)1024 * 64 * 2);
  _Float16* Wff1  = (_Float16*)alloc((size_t)2048 * 1024 * 2);
  _Float16* Wff3  = (_Float16*)alloc((size_t)2048 * 1024 * 2);
  _Float16* Wff2  = (_Float16*)alloc((size_t)1024 * 2048 * 2);
  _Float16* x16   = (_Float16*)alloc((size_t)MM * DD * 2);
  _Float16* h16   = (_Float16*)alloc((size_t)MM * DD * 2);   // reused later as yn16
  float*    hn32  = (float*)alloc((size_t)MM * DD * 4);      // reused later as attn
  float*    qkvs  = (float*)alloc((size_t)MM * 2048 * 4);    // reused: [G | Y1]
  float*    amid32= (float*)alloc((size_t)MM * 64 * 4);      // reused as gate mid
  _Float16* amid16= (_Float16*)alloc((size_t)MM * 64 * 2);
  float*    abase = (float*)alloc((size_t)MM * 512 * 4);     // alpha base -> alpha gate
  float*    bbase = (float*)alloc((size_t)MM * 8 * 4);
  float*    betau = (float*)alloc((size_t)MM * 8 * 4);
  float*    ybuf  = (float*)alloc((size_t)MM * 512 * 4);
  _Float16* yhn16 = (_Float16*)alloc((size_t)MM * 512 * 2);
  float*    F1    = (float*)alloc((size_t)MM * 2048 * 4);    // reused for ff2 out
  float*    F3    = (float*)alloc((size_t)MM * 2048 * 4);
  _Float16* ffh16 = (_Float16*)alloc((size_t)MM * 2048 * 2);

  float* attn = hn32;                 // hn dead after conv
  float* G    = qkvs;                 // qkvs dead after scan
  float* Y1   = qkvs + (size_t)MM * 1024;
  _Float16* yn16 = h16;               // h16 dead after alpha_up GEMM
  float* gmid32 = amid32;
  _Float16* gmid16 = amid16;
  float* ff2out = F1;

  auto cvt = [&](const float* src, _Float16* dst, size_t n) {
    f32_to_f16_kernel<<<2048, 256, 0, stream>>>(src, dst, n);
  };
  auto gemm = [&](const _Float16* A, const _Float16* W, const float* bias,
                  float* Cf, _Float16* Ch, int Mn, int Nn, int Kn, int act) {
    dim3 g(Nn / 64, Mn / 64);
    gemm_wmma_kernel<<<g, 128, 0, stream>>>(A, W, bias, Cf, Ch, Mn, Nn, Kn, act);
  };

  // ---- weight conversions ----
  cvt(q_w,        Wqkvs + (size_t)0   * 1024, (size_t)512 * 1024);
  cvt(k_w,        Wqkvs + (size_t)512 * 1024, (size_t)512 * 1024);
  cvt(v_w,        Wqkvs + (size_t)1024* 1024, (size_t)512 * 1024);
  cvt(spike_w,    Wqkvs + (size_t)1536* 1024, (size_t)512 * 1024);
  cvt(alpha_up_w, Wau,  (size_t)64 * 1024);
  cvt(alpha_dn_w, Wad,  (size_t)512 * 64);
  cvt(out_w,      Wout, (size_t)1024 * 512);
  cvt(u1_w,       Wu1,  (size_t)64 * 1024);
  cvt(u2_w,       Wu2,  (size_t)1024 * 64);
  cvt(ff_w1,      Wff1, (size_t)2048 * 1024);
  cvt(ff_w3,      Wff3, (size_t)2048 * 1024);
  cvt(ff_w2,      Wff2, (size_t)1024 * 2048);
  cvt(x,          x16,  (size_t)MM * DD);

  // ---- 1. input RMSNorm -> hn (fp32 for conv) ----
  rms_row_kernel<<<MM, 256, 0, stream>>>(x, norm_in_w, hn32, nullptr, DD);
  // ---- 2. causal depthwise conv + SiLU -> h (fp16) ----
  conv_silu_kernel<<<MM, 256, 0, stream>>>(hn32, conv_w, conv_b, h16);
  // ---- 3. fused q|k|v|spike projection (WMMA) ----
  gemm(h16, Wqkvs, nullptr, qkvs, nullptr, MM, 2048, 1024, 0);
  // ---- 4. per-head RMS on q,k,v (in-place) ----
  qkv_rms_kernel<<<3 * MM * HH, 64, 0, stream>>>(qkvs, q_nw, k_nw, v_nw);
  // ---- 5. beta base (tiny N=8) ----
  beta_kernel<<<(MM * HH + 255) / 256, 256, 0, stream>>>(h16, beta_w, beta_b, bbase);
  // ---- 6. alpha path: silu(h@up^T + b) @ down^T + b (WMMA x2) ----
  gemm(h16, Wau, alpha_up_b, amid32, amid16, MM, 64, 1024, 1);
  gemm(amid16, Wad, alpha_dn_b, abase, nullptr, MM, 512, 64, 0);
  // ---- 7. LIF spike chain -> alpha gate (in-place), u = active*beta ----
  spike_kernel<<<BB * HH, 64, 0, stream>>>(qkvs, abase, bbase, alpha_sp, beta_sp, betau);
  // ---- 8. gated delta-rule scan (LDS state) ----
  scan_kernel<<<BB * HH, 256, 0, stream>>>(qkvs, abase, betau, ybuf);
  // ---- 9. head RMS -> fp16 ----
  head_rms_kernel<<<MM * HH, 64, 0, stream>>>(ybuf, head_nw, yhn16);
  // ---- 10. out projection (WMMA) ----
  gemm(yhn16, Wout, nullptr, attn, nullptr, MM, 1024, 512, 0);
  // ---- 11. gate: sigmoid(silu(x@u1^T+b)@u2^T+b)  (WMMA x2, fused act) ----
  gemm(x16, Wu1, u1_b, gmid32, gmid16, MM, 64, 1024, 1);
  gemm(gmid16, Wu2, u2_b, G, nullptr, MM, 1024, 64, 2);
  // ---- 12. y1 = x + attn * gate ----
  fuse_gate_kernel<<<4096, 256, 0, stream>>>(x, attn, G, Y1, (size_t)MM * DD);
  // ---- 13. FFN: rms -> swiglu (WMMA x3) -> residual ----
  rms_row_kernel<<<MM, 256, 0, stream>>>(Y1, ff_norm_w, nullptr, yn16, DD);
  gemm(yn16, Wff1, nullptr, F1, nullptr, MM, 2048, 1024, 1);  // silu fused
  gemm(yn16, Wff3, nullptr, F3, nullptr, MM, 2048, 1024, 0);
  fuse_ff_kernel<<<4096, 256, 0, stream>>>(F1, F3, ffh16, (size_t)MM * DFFF);
  gemm(ffh16, Wff2, nullptr, ff2out, nullptr, MM, 1024, 2048, 0);
  final_add_kernel<<<4096, 256, 0, stream>>>(Y1, ff2out, outp, (size_t)MM * DD);
}
